// GCNWithGATAndTCN_58437325029914
// MI455X (gfx1250) — matrix operand order repaired and checked by reading, not compiled, not run on hardware
//
// ============================================================================
// GCN + GAT + TCN + NonLocal pipeline, AMD MI455X (gfx1250, CDNA5, wave32).
//
// ~88 GFLOP of GEMMs vs ~1.2 GB HBM traffic (23.3 TB/s) -> matrix-core bound;
// all GEMMs use v_wmma_f32_16x16x32_bf16 (bf16 operands, f32 accumulate).
//
// GEMM design:
//  * NT-only: C[m,n] = sum_k A[m,k] * Bt[n,k]. Weights pre-transposed once;
//    big per-batch B operands (support, g) are written transposed directly by
//    the producing GEMM's epilogue (one 16B store per accumulator).
//  * Block tile 128x128, 8 waves in 4x2, each wave owns 32x64 of C:
//    8 WMMA accumulators fed by 2 A frags x 4 B frags = 12 ds_load_b128 per
//    8 WMMAs (1.5 LDS ops/WMMA vs 2.0 at 32x32) - LDS bandwidth per matrix
//    op is the limiter once WMMA rate is the ceiling.
//  * Double-buffered ASYNC staging: 4x global_load_async_to_lds_b128 per
//    thread per K-tile (A tile 128x32 + B tile 128x32, 2x16B chunks each);
//    s_wait_asynccnt 0x4 keeps tile k+1's DMA in flight under tile k's WMMAs
//    (ASYNCcnt is per-wave, in-order, 4 ops/tile). global_prefetch_b8 two
//    tiles ahead.
//  * Fragments are two ds_load_b128 each, matching the ISA VGPR layouts
//    (CDNA5 ISA 7.12.2): A lane elems 0..7 -> k=laneHi*8+0..7,
//    8..15 -> k=16+laneHi*8+0..7; B lane elems: same k order as A.
//  * Epilogue fuses bias / f32 residual / (Leaky)ReLU / f32+bf16(+bf16^T).
// Workspace ~282 MB; the 134MB NxN score buffer is reused for im2col and the
// in-place bf16 row-softmax overlay.
// ============================================================================
#include <hip/hip_runtime.h>
#include <hip/hip_bf16.h>

typedef __attribute__((ext_vector_type(16))) __bf16 v16bf;
typedef __attribute__((ext_vector_type(8)))  float  v8f;

#define kB  8
#define kN  2048
#define kNF 128
#define kNH 256
#define kNO 256

// ---- CDNA5 async global->LDS copy (16B per lane) + ASYNCcnt waits --------
__device__ __forceinline__ void async_b128(unsigned ldsAddr, const void* gaddr) {
#if defined(__HIP_DEVICE_COMPILE__)
  asm volatile("global_load_async_to_lds_b128 %0, %1, off"
               :: "v"(ldsAddr), "v"((unsigned long long)gaddr) : "memory");
#else
  (void)ldsAddr; (void)gaddr;
#endif
}
__device__ __forceinline__ void wait_async_le4() {
#if defined(__HIP_DEVICE_COMPILE__)
  asm volatile("s_wait_asynccnt 0x4" ::: "memory");
#endif
}
__device__ __forceinline__ void wait_async_0() {
#if defined(__HIP_DEVICE_COMPILE__)
  asm volatile("s_wait_asynccnt 0x0" ::: "memory");
#endif
}

// ---------------------------------------------------------------------------
// Batched NT GEMM: C[M,Nc] = A[M,K] * Bt[Nc,K]^T  (bf16 in, f32 accum).
// Optional bias[n], f32 residual, act (0 none, 1 relu, 2 leaky 0.2),
// f32 / bf16 / transposed-bf16 outputs. All dims divisible by tile sizes.
// ---------------------------------------------------------------------------
__global__ __launch_bounds__(256) void k_gemm_bf16_wmma(
    const __bf16* __restrict__ A, long long sA,
    const __bf16* __restrict__ Bt, long long sB,
    int M, int Nc, int K, int lda, int ldb, int ldc,
    const float* __restrict__ bias,
    const float* __restrict__ resid, long long sR, int act,
    float* __restrict__ outF, long long sOF,
    __bf16* __restrict__ outH, long long sOH,
    __bf16* __restrict__ outT, long long sOT, int ldt)
{
  __shared__ __align__(16) __bf16 As[2][128 * 40];   // [m][k], stride 40 (80B)
  __shared__ __align__(16) __bf16 Bs[2][128 * 40];   // [n][k], stride 40

  const int tid    = threadIdx.x;
  const int lane   = tid & 31;
  const int wave   = tid >> 5;
  const int laneHi = (lane >> 4) & 1;
  const int l15    = lane & 15;

  const int row0 = blockIdx.y * 128;
  const int col0 = blockIdx.x * 128;
  const long long bz = blockIdx.z;

  const __bf16* Ab = A  + bz * sA;
  const __bf16* Bb = Bt + bz * sB;

  const int wm = (wave >> 1) * 32;   // wave M offset in tile: 0/32/64/96
  const int wn = (wave & 1) * 64;    // wave N offset in tile: 0/64

  v8f acc[8] = {};                   // [mi(2)][ni(4)]

  const int sRow = tid >> 1;         // staging: 128 rows x 2 thr, 2x16B each
  const int sCol = (tid & 1) * 16;
  const unsigned asB[2] = { (unsigned)(size_t)&As[0][0], (unsigned)(size_t)&As[1][0] };
  const unsigned bsB[2] = { (unsigned)(size_t)&Bs[0][0], (unsigned)(size_t)&Bs[1][0] };
  const unsigned sLds = (unsigned)(sRow * 40 + sCol) * 2u;

  const int kTiles = K / 32;

  auto stage = [&](int buf, int kt) {
    const int k0 = kt * 32;
    const __bf16* aSrc = Ab + (long long)(row0 + sRow) * lda + (k0 + sCol);
    const __bf16* bSrc = Bb + (long long)(col0 + sRow) * ldb + (k0 + sCol);
    async_b128(asB[buf] + sLds,       aSrc);
    async_b128(asB[buf] + sLds + 16u, aSrc + 8);
    async_b128(bsB[buf] + sLds,       bSrc);
    async_b128(bsB[buf] + sLds + 16u, bSrc + 8);
    if (k0 + 96 <= K) {  // global_prefetch_b8 two tiles ahead
      __builtin_prefetch(aSrc + 64, 0, 3);
      __builtin_prefetch(bSrc + 64, 0, 3);
    }
  };

  stage(0, 0);
  if (kTiles > 1) stage(1, 1);

  for (int kt = 0; kt < kTiles; ++kt) {
    const int cur = kt & 1;
    if (kt + 1 < kTiles) wait_async_le4();   // tile kt done, kt+1 in flight
    else                 wait_async_0();
    __syncthreads();

    const __bf16* Ac = &As[cur][0];
    const __bf16* Bc = &Bs[cur][0];
    union Frag { v16bf v; uint4 q[2]; };
    Frag a0, a1, b[4];
    {
      const int ra0 = (wm + l15) * 40;
      const int ra1 = (wm + 16 + l15) * 40;
      a0.q[0] = *(const uint4*)(Ac + ra0 + laneHi * 8);
      a0.q[1] = *(const uint4*)(Ac + ra0 + 16 + laneHi * 8);
      a1.q[0] = *(const uint4*)(Ac + ra1 + laneHi * 8);
      a1.q[1] = *(const uint4*)(Ac + ra1 + 16 + laneHi * 8);
#pragma unroll
      for (int j = 0; j < 4; ++j) {
        const int rb = (wn + j * 16 + l15) * 40 + laneHi * 8;
        b[j].q[0] = *(const uint4*)(Bc + rb);
        b[j].q[1] = *(const uint4*)(Bc + rb + 16);
      }
    }
#if defined(__HIP_DEVICE_COMPILE__)
#pragma unroll
    for (int j = 0; j < 4; ++j) {
      acc[j]     = __builtin_amdgcn_wmma_f32_16x16x32_bf16(false, a0.v, false, b[j].v, (short)0, acc[j],     false, false);
      acc[4 + j] = __builtin_amdgcn_wmma_f32_16x16x32_bf16(false, a1.v, false, b[j].v, (short)0, acc[4 + j], false, false);
    }
#else
    (void)a0; (void)a1; (void)b;
#endif
    __syncthreads();                         // all waves done reading buf cur
    if (kt + 2 < kTiles) stage(cur, kt + 2); // refill freed buffer
  }

  float* oF = outF ? outF + bz * sOF : nullptr;
  __bf16* oH = outH ? outH + bz * sOH : nullptr;
  __bf16* oT = outT ? outT + bz * sOT : nullptr;
  const float* rs = resid ? resid + bz * sR : nullptr;

#pragma unroll
  for (int mi = 0; mi < 2; ++mi) {
#pragma unroll
    for (int ni = 0; ni < 4; ++ni) {
      const v8f a = acc[mi * 4 + ni];
      // C layout: accum elem r -> m = r + 8*laneHi, n = l15 (per 16x16 tile)
      const int mBase = row0 + wm + mi * 16 + laneHi * 8;
      const int nCol  = col0 + wn + ni * 16 + l15;
      const float bi = bias ? bias[nCol] : 0.0f;
      float v[8];
#pragma unroll
      for (int r = 0; r < 8; ++r) {
        const long long idx = (long long)(mBase + r) * ldc + nCol;
        float w = a[r] + bi;
        if (rs)  w += rs[idx];
        if (act == 1)      w = w > 0.0f ? w : 0.0f;
        else if (act == 2) w = w > 0.0f ? w : 0.2f * w;   // LeakyReLU(0.2)
        v[r] = w;
        if (oF) oF[idx] = w;
        if (oH) oH[idx] = (__bf16)w;
      }
      if (oT) {  // transposed bf16 store: 8 consecutive M -> one 16B write
        union { __bf16 h[8]; uint4 q; } p;
#pragma unroll
        for (int r = 0; r < 8; ++r) p.h[r] = (__bf16)v[r];
        *(uint4*)(oT + (long long)nCol * ldt + mBase) = p.q;
      }
    }
  }
}

// ------------------------------- helpers -----------------------------------
__global__ void k_f32_to_bf16(const float* __restrict__ in, __bf16* __restrict__ out, long long n) {
  long long i = (long long)blockIdx.x * 256 + threadIdx.x;
  if (i < n) out[i] = (__bf16)in[i];
}

// W[R][C] f32 -> Wt[C][R] bf16 (grid R blocks, C threads)
__global__ void k_transpose_cvt(const float* __restrict__ W, __bf16* __restrict__ Wt, int R, int C) {
  const int r = blockIdx.x, c = threadIdx.x;
  Wt[c * R + r] = (__bf16)W[r * C + c];
}

// tcn_W[O=256][I=256][K=3] -> W2t[o][dk*256+i] bf16 (Bt for im2col GEMM)
__global__ void k_pack_tcn_w(const float* __restrict__ W, __bf16* __restrict__ W2t) {
  const int o = blockIdx.x, i = threadIdx.x;
#pragma unroll
  for (int dk = 0; dk < 3; ++dk)
    W2t[o * 768 + dk * 256 + i] = (__bf16)W[(o * 256 + i) * 3 + dk];
}

__global__ void k_zero(float* __restrict__ p) { p[blockIdx.x * 256 + threadIdx.x] = 0.0f; }

// Per-channel sum/sumsq over all B*N=16384 rows (coalesced, atomic partials)
__global__ void k_bn_stats(const float* __restrict__ h, float* __restrict__ sum,
                           float* __restrict__ sumsq) {
  const int c = threadIdx.x;
  const long long r0 = (long long)blockIdx.x * 64;
  float s = 0.0f, s2 = 0.0f;
  for (int r = 0; r < 64; ++r) {
    const float v = h[(r0 + r) * 256 + c];
    s += v; s2 += v * v;
  }
  atomicAdd(&sum[c], s);
  atomicAdd(&sumsq[c], s2);
}

__global__ void k_bn1_apply(const float* __restrict__ h, const float* __restrict__ sum,
                            const float* __restrict__ sumsq, const float* __restrict__ g,
                            const float* __restrict__ b, __bf16* __restrict__ out) {
  const int c = threadIdx.x;
  const long long r = blockIdx.x;
  const float inv = 1.0f / 16384.0f;
  const float m = sum[c] * inv;
  const float var = sumsq[c] * inv - m * m;
  float v = (h[r * 256 + c] - m) * rsqrtf(var + 1e-5f) * g[c] + b[c];
  v = v > 0.0f ? v : 0.0f;
  out[r * 256 + c] = (__bf16)v;
}

// y = relu(relu(bn(conv)) + h2); write f32 + bf16
__global__ void k_tcn_bn_res(const float* __restrict__ conv, const float* __restrict__ sum,
                             const float* __restrict__ sumsq, const float* __restrict__ g,
                             const float* __restrict__ b, const float* __restrict__ h2,
                             float* __restrict__ yf, __bf16* __restrict__ ybf) {
  const int c = threadIdx.x;
  const long long r = blockIdx.x;
  const float inv = 1.0f / 16384.0f;
  const float m = sum[c] * inv;
  const float var = sumsq[c] * inv - m * m;
  float v = (conv[r * 256 + c] - m) * rsqrtf(var + 1e-5f) * g[c] + b[c];
  v = v > 0.0f ? v : 0.0f;
  v += h2[r * 256 + c];
  v = v > 0.0f ? v : 0.0f;
  yf[r * 256 + c] = v;
  ybf[r * 256 + c] = (__bf16)v;
}

// GAT softmax(dim=1): online max/sumexp down each column m of E[b,n,m]
__global__ void k_gat_colstats(const float* __restrict__ E, float* __restrict__ colmax,
                               float* __restrict__ colsum) {
  const int m = blockIdx.x * 256 + threadIdx.x;
  const long long b = blockIdx.y;
  const float* Eb = E + b * (long long)kN * kN;
  float mx = -3.0e38f, s = 0.0f;
  for (int n = 0; n < kN; ++n) {
    const float v = Eb[(long long)n * kN + m];
    const float nm = fmaxf(mx, v);
    s = s * __expf(mx - nm) + __expf(v - nm);
    mx = nm;
  }
  colmax[b * kN + m] = mx;
  colsum[b * kN + m] = s;
}

// s[b,n] = sum_m exp(E[b,n,m]-colmax[m]) / colsum[m]   (== att.sum(axis=2))
__global__ void k_gat_rowsum(const float* __restrict__ E, const float* __restrict__ colmax,
                             const float* __restrict__ colsum, float* __restrict__ sarr) {
  __shared__ float red[256];
  const long long b = blockIdx.y;
  const int n = blockIdx.x;
  const float* row = E + (b * kN + n) * (long long)kN;
  const float* cm = colmax + b * kN;
  const float* cs = colsum + b * kN;
  float acc = 0.0f;
  for (int m = threadIdx.x; m < kN; m += 256)
    acc += __expf(row[m] - cm[m]) / cs[m];
  red[threadIdx.x] = acc;
  __syncthreads();
  for (int o = 128; o > 0; o >>= 1) {
    if (threadIdx.x < o) red[threadIdx.x] += red[threadIdx.x + o];
    __syncthreads();
  }
  if (threadIdx.x == 0) sarr[b * kN + n] = red[0];
}

__global__ void k_gat_scale(const float* __restrict__ hp, const float* __restrict__ sarr,
                            float* __restrict__ h2) {
  const long long r = blockIdx.x;
  const int c = threadIdx.x;
  h2[r * 256 + c] = hp[r * 256 + c] * sarr[r];
}

// im2col K=3 with zero pad: A2[b][n][dk*256+i] = h2[b][n-1+dk][i]
__global__ void k_im2col3(const float* __restrict__ h2, __bf16* __restrict__ A2) {
  const int i = threadIdx.x, dk = blockIdx.x, n = blockIdx.y;
  const long long b = blockIdx.z;
  const int srcn = n - 1 + dk;
  const float v = (srcn >= 0 && srcn < kN) ? h2[(b * kN + srcn) * 256 + i] : 0.0f;
  A2[(b * kN + n) * 768 + dk * 256 + i] = (__bf16)v;
}

// NonLocal row softmax; writes bf16 result in place over the f32 row start
// (bf16 row stride = 2*kN elems => rows never overlap).
__global__ __launch_bounds__(256) void k_row_softmax_bf16(float* __restrict__ E) {
  __shared__ float buf[kN];
  __shared__ float red[256];
  const int tid = threadIdx.x;
  float* row = E + ((long long)blockIdx.y * kN + blockIdx.x) * kN;
  float mx = -3.0e38f;
  for (int m = tid; m < kN; m += 256) { const float v = row[m]; buf[m] = v; mx = fmaxf(mx, v); }
  red[tid] = mx;
  __syncthreads();
  for (int o = 128; o > 0; o >>= 1) {
    if (tid < o) red[tid] = fmaxf(red[tid], red[tid + o]);
    __syncthreads();
  }
  mx = red[0];
  __syncthreads();
  float s = 0.0f;
  for (int m = tid; m < kN; m += 256) s += __expf(buf[m] - mx);
  red[tid] = s;
  __syncthreads();
  for (int o = 128; o > 0; o >>= 1) {
    if (tid < o) red[tid] += red[tid + o];
    __syncthreads();
  }
  const float inv = 1.0f / red[0];
  __bf16* orow = (__bf16*)row;
  for (int m = tid; m < kN; m += 256) orow[m] = (__bf16)(__expf(buf[m] - mx) * inv);
}

// ---------------------------------------------------------------------------
extern "C" void kernel_launch(void* const* d_in, const int* in_sizes, int n_in,
                              void* d_out, int out_size, void* d_ws, size_t ws_size,
                              hipStream_t stream) {
  (void)in_sizes; (void)n_in; (void)out_size; (void)ws_size;
  const float* x        = (const float*)d_in[0];
  /* d_in[1] adj: unused by reference math */
  const float* adj_gcn  = (const float*)d_in[2];
  const float* gcn_W    = (const float*)d_in[3];
  const float* gcn_b    = (const float*)d_in[4];
  const float* gcn_resW = (const float*)d_in[5];
  const float* bn1_g    = (const float*)d_in[6];
  const float* bn1_b    = (const float*)d_in[7];
  const float* gat_W    = (const float*)d_in[8];
  const float* tcn_W    = (const float*)d_in[9];
  const float* tcn_b    = (const float*)d_in[10];
  const float* tbn_g    = (const float*)d_in[11];
  const float* tbn_b    = (const float*)d_in[12];
  const float* theta_W  = (const float*)d_in[13];
  const float* theta_b  = (const float*)d_in[14];
  const float* phi_W    = (const float*)d_in[15];
  const float* phi_b    = (const float*)d_in[16];
  const float* g_W      = (const float*)d_in[17];
  const float* g_b      = (const float*)d_in[18];
  const float* nl_W     = (const float*)d_in[19];
  const float* nl_b     = (const float*)d_in[20];

  char* base = (char*)d_ws;
  size_t off = 0;
  auto alloc = [&](size_t bytes) -> char* {
    char* p = base + off;
    off += (bytes + 255) & ~(size_t)255;
    return p;
  };

  float*  E      = (float*) alloc((size_t)kB * kN * kN * 4);   // 134 MB scores
  __bf16* A2bf   = (__bf16*)E;   // im2col buffer, disjoint lifetime with E
  __bf16* att2   = (__bf16*)E;   // bf16 softmax overlay (row stride 2*kN)
  __bf16* xbf    = (__bf16*)alloc((size_t)kB * kN * kNF * 2);
  __bf16* adjbf  = (__bf16*)alloc((size_t)kN * kN * 2);
  __bf16* gcnWt  = (__bf16*)alloc((size_t)kNH * kNF * 2);   // [NH][NF]
  __bf16* grWt   = (__bf16*)alloc((size_t)kNH * kNF * 2);
  __bf16* gatWt  = (__bf16*)alloc((size_t)kNO * kNH * 2);   // [NO][NH]
  __bf16* thWbf  = (__bf16*)alloc((size_t)kNO * kNO * 2);   // [O][C] as-is = Bt
  __bf16* phWbf  = (__bf16*)alloc((size_t)kNO * kNO * 2);
  __bf16* gWbf   = (__bf16*)alloc((size_t)kNO * kNO * 2);
  __bf16* nlWbf  = (__bf16*)alloc((size_t)kNO * kNO * 2);
  __bf16* W2t    = (__bf16*)alloc((size_t)kNO * 768 * 2);   // [O][3*C]
  __bf16* supT   = (__bf16*)alloc((size_t)kB * kN * kNH * 2); // support^T [NH][N]
  float*  resf   = (float*) alloc((size_t)kB * kN * kNH * 4);
  float*  hf     = (float*) alloc((size_t)kB * kN * kNH * 4);
  __bf16* hnbf   = (__bf16*)alloc((size_t)kB * kN * kNH * 2);
  float*  hpf    = (float*) alloc((size_t)kB * kN * kNO * 4);
  __bf16* hpbf   = (__bf16*)alloc((size_t)kB * kN * kNO * 2);
  float*  colmax = (float*) alloc((size_t)kB * kN * 4);
  float*  colsum = (float*) alloc((size_t)kB * kN * 4);
  float*  sarr   = (float*) alloc((size_t)kB * kN * 4);
  float*  h2f    = resf;  // res dead after GCN combine
  float*  convf  = hf;    // h dead after bn1 apply
  float*  yf     = (float*) alloc((size_t)kB * kN * kNO * 4);
  __bf16* ybf    = (__bf16*)alloc((size_t)kB * kN * kNO * 2);
  __bf16* thbf   = (__bf16*)alloc((size_t)kB * kN * kNO * 2);
  __bf16* phbf   = (__bf16*)alloc((size_t)kB * kN * kNO * 2);
  __bf16* ggT    = (__bf16*)alloc((size_t)kB * kN * kNO * 2); // g^T [NO][N]
  __bf16* obf    = (__bf16*)alloc((size_t)kB * kN * kNO * 2);
  float*  bnsum  = (float*) alloc(512 * 4);

  auto gemm = [&](const __bf16* A, long long sA, const __bf16* Bp, long long sB,
                  int M, int Nc, int K, int lda, int ldb, int ldc,
                  const float* bias, const float* resid, long long sR, int act,
                  float* oF, long long sOF, __bf16* oH, long long sOH,
                  __bf16* oT, long long sOT, int ldt) {
    dim3 grid(Nc / 128, M / 128, kB);
    k_gemm_bf16_wmma<<<grid, 256, 0, stream>>>(A, sA, Bp, sB, M, Nc, K, lda, ldb, ldc,
                                               bias, resid, sR, act,
                                               oF, sOF, oH, sOH, oT, sOT, ldt);
  };

  // ---- precision staging (f32 -> bf16 operands, weights transposed) ----
  long long nx = (long long)kB * kN * kNF;
  k_f32_to_bf16<<<dim3((unsigned)((nx + 255) / 256)), 256, 0, stream>>>(x, xbf, nx);
  long long na = (long long)kN * kN;
  k_f32_to_bf16<<<dim3((unsigned)((na + 255) / 256)), 256, 0, stream>>>(adj_gcn, adjbf, na);
  k_transpose_cvt<<<kNF, kNH, 0, stream>>>(gcn_W, gcnWt, kNF, kNH);
  k_transpose_cvt<<<kNF, kNH, 0, stream>>>(gcn_resW, grWt, kNF, kNH);
  k_transpose_cvt<<<kNH, kNO, 0, stream>>>(gat_W, gatWt, kNH, kNO);
  k_f32_to_bf16<<<(kNO * kNO) / 256, 256, 0, stream>>>(theta_W, thWbf, kNO * kNO);
  k_f32_to_bf16<<<(kNO * kNO) / 256, 256, 0, stream>>>(phi_W, phWbf, kNO * kNO);
  k_f32_to_bf16<<<(kNO * kNO) / 256, 256, 0, stream>>>(g_W, gWbf, kNO * kNO);
  k_f32_to_bf16<<<(kNO * kNO) / 256, 256, 0, stream>>>(nl_W, nlWbf, kNO * kNO);
  k_pack_tcn_w<<<256, 256, 0, stream>>>(tcn_W, W2t);

  const long long sX = (long long)kN * kNF;
  const long long sC = (long long)kN * kNH;   // == kN*kNO
  const long long sE = (long long)kN * kN;

  // ---- GCN ----
  gemm(xbf, sX, gcnWt, 0, kN, kNH, kNF, kNF, kNF, kNH,
       nullptr, nullptr, 0, 0, nullptr, 0, nullptr, 0, supT, sC, kN);      // support^T
  gemm(xbf, sX, grWt, 0, kN, kNH, kNF, kNF, kNF, kNH,
       nullptr, nullptr, 0, 0, resf, sC, nullptr, 0, nullptr, 0, 0);       // residual
  gemm(adjbf, 0, supT, sC, kN, kNH, kN, kN, kN, kNH,
       gcn_b, resf, sC, 1, hf, sC, nullptr, 0, nullptr, 0, 0);  // h=relu(adj@sup+b+res)

  // ---- bn1 + relu ----
  k_zero<<<2, 256, 0, stream>>>(bnsum);
  k_bn_stats<<<(kB * kN) / 64, 256, 0, stream>>>(hf, bnsum, bnsum + 256);
  k_bn1_apply<<<kB * kN, 256, 0, stream>>>(hf, bnsum, bnsum + 256, bn1_g, bn1_b, hnbf);

  // ---- GAT ----
  gemm(hnbf, sC, gatWt, 0, kN, kNO, kNH, kNH, kNH, kNO,
       nullptr, nullptr, 0, 0, hpf, sC, hpbf, sC, nullptr, 0, 0);          // hp
  gemm(hpbf, sC, hpbf, sC, kN, kN, kNO, kNO, kNO, kN,
       nullptr, nullptr, 0, 2, E, sE, nullptr, 0, nullptr, 0, 0);  // e=leaky(hp hp^T)
  k_gat_colstats<<<dim3(kN / 256, kB), 256, 0, stream>>>(E, colmax, colsum);
  k_gat_rowsum<<<dim3(kN, kB), 256, 0, stream>>>(E, colmax, colsum, sarr);
  k_gat_scale<<<kB * kN, 256, 0, stream>>>(hpf, sarr, h2f);   // h2 = hp * att.sum(2)

  // ---- TCN (im2col + GEMM + bn + residual relu) ----
  k_im2col3<<<dim3(3, kN, kB), 256, 0, stream>>>(h2f, A2bf);
  gemm(A2bf, (long long)kN * 768, W2t, 0, kN, kNO, 768, 768, 768, kNO,
       tcn_b, nullptr, 0, 0, convf, sC, nullptr, 0, nullptr, 0, 0);
  k_zero<<<2, 256, 0, stream>>>(bnsum);
  k_bn_stats<<<(kB * kN) / 64, 256, 0, stream>>>(convf, bnsum, bnsum + 256);
  k_tcn_bn_res<<<kB * kN, 256, 0, stream>>>(convf, bnsum, bnsum + 256, tbn_g, tbn_b,
                                            h2f, yf, ybf);

  // ---- NonLocal ----
  gemm(ybf, sC, thWbf, 0, kN, kNO, kNO, kNO, kNO, kNO,
       theta_b, nullptr, 0, 0, nullptr, 0, thbf, sC, nullptr, 0, 0);       // theta
  gemm(ybf, sC, phWbf, 0, kN, kNO, kNO, kNO, kNO, kNO,
       phi_b, nullptr, 0, 0, nullptr, 0, phbf, sC, nullptr, 0, 0);         // phi
  gemm(ybf, sC, gWbf, 0, kN, kNO, kNO, kNO, kNO, kNO,
       g_b, nullptr, 0, 0, nullptr, 0, nullptr, 0, ggT, sC, kN);           // g^T
  gemm(thbf, sC, phbf, sC, kN, kN, kNO, kNO, kNO, kN,
       nullptr, nullptr, 0, 0, E, sE, nullptr, 0, nullptr, 0, 0);  // f = th ph^T
  k_row_softmax_bf16<<<dim3(kN, kB), 256, 0, stream>>>(E);         // att2 (bf16 overlay)
  gemm(att2, 2LL * sE, ggT, sC, kN, kNO, kN, 2 * kN, kN, kNO,
       nullptr, nullptr, 0, 0, nullptr, 0, obf, sC, nullptr, 0, 0);        // o = att2 g
  gemm(obf, sC, nlWbf, 0, kN, kNO, kNO, kNO, kNO, kNO,
       nl_b, yf, sC, 0, (float*)d_out, sC, nullptr, 0, nullptr, 0, 0);     // out
}
